// LSTMCell_32744830664921
// MI455X (gfx1250) — matrix operand order repaired and checked by reading, not compiled, not run on hardware
//
#include <hip/hip_runtime.h>
#include <hip/hip_bf16.h>

typedef __attribute__((ext_vector_type(16))) _Float16 v16h;
typedef __attribute__((ext_vector_type(8)))  _Float16 v8h;
typedef __attribute__((ext_vector_type(4)))  _Float16 v4h;
typedef __attribute__((ext_vector_type(8)))  float    v8f;

#define B_SZ 8192
#define H_SZ 1024
#define K_SZ 1024

// ---------------- fp32 -> f16 conversion ----------------
__global__ void cvt_f32_f16(const float* __restrict__ s, _Float16* __restrict__ d, int n4) {
    int i = blockIdx.x * blockDim.x + threadIdx.x;
    if (i < n4) {
        float4 v = ((const float4*)s)[i];
        v4h h = { (_Float16)v.x, (_Float16)v.y, (_Float16)v.z, (_Float16)v.w };
        ((v4h*)d)[i] = h;
    }
}

// ---------------- WMMA helpers ----------------
// A tile (16x32 f16, MxK): lane l<16 holds row M=l, K = {k0..k0+7, k0+16..k0+23};
// lane l>=16 holds row M=l-16, K = {k0+8..k0+15, k0+24..k0+31}.  (ISA 7.12.2)
__device__ __forceinline__ v16h loadA(const _Float16* p) {
    v8h lo = *(const v8h*)(p);
    v8h hi = *(const v8h*)(p + 16);
    return __builtin_shufflevector(lo, hi, 0,1,2,3,4,5,6,7,8,9,10,11,12,13,14,15);
}
// B tile (32x16 f16, KxN): lane l<16 holds col N=l, K = {k0..k0+7, k0+16..k0+23};
// lane l>=16 holds col N=l-16, K = {k0+8..k0+15, k0+24..k0+31}.
__device__ __forceinline__ v16h loadB(const _Float16* p) {
    v8h lo = *(const v8h*)(p);
    v8h hi = *(const v8h*)(p + 16);
    return __builtin_shufflevector(lo, hi, 0,1,2,3,4,5,6,7,8,9,10,11,12,13,14,15);
}
__device__ __forceinline__ v8f wmma_f16(v16h a, v16h b, v8f c) {
    return __builtin_amdgcn_wmma_f32_16x16x32_f16(false, a, false, b, (short)0, c, false, false);
}
__device__ __forceinline__ float sigm(float x) {
    return 1.0f / (1.0f + __expf(-x));
}

// ---------------- fused LSTM+attention cell ----------------
// Each wave: 32 batch rows x 16 hidden cols; 6 logical GEMM accumulators
// (ingate, forget, cell, out from [input|hx]@[Wih|Whh]^T; ingate_att, attgate
// from att@Watt^T), epilogue fused.
__global__ void __launch_bounds__(256)
lstm_att_wmma(const _Float16* __restrict__ X,  const _Float16* __restrict__ HX,
              const _Float16* __restrict__ AT,
              const _Float16* __restrict__ WIH, const _Float16* __restrict__ WHH,
              const _Float16* __restrict__ WAT,
              const float* __restrict__ bih, const float* __restrict__ bhh,
              const float* __restrict__ batt, const float* __restrict__ cx,
              float* __restrict__ out)
{
    const int lane = threadIdx.x & 31;
    const int wave = threadIdx.x >> 5;
    const int flat = blockIdx.x * 8 + wave;          // 16384 wave-tiles total
    const int bm   = (flat >> 6) * 32;               // batch tile base (256 tiles)
    const int hn   = (flat & 63) * 16;               // hidden tile base (64 tiles)

    const int lrow   = lane & 15;
    const int kA     = (lane >> 4) * 8;              // A-lane K split
    const int kB     = (lane >> 4) * 8;              // B-lane K split

    // A row base pointers (per-lane)
    const _Float16* xr0 = X  + (size_t)(bm      + lrow) * K_SZ + kA;
    const _Float16* xr1 = X  + (size_t)(bm + 16 + lrow) * K_SZ + kA;
    const _Float16* hr0 = HX + (size_t)(bm      + lrow) * K_SZ + kA;
    const _Float16* hr1 = HX + (size_t)(bm + 16 + lrow) * K_SZ + kA;
    const _Float16* ar0 = AT + (size_t)(bm      + lrow) * K_SZ + kA;
    const _Float16* ar1 = AT + (size_t)(bm + 16 + lrow) * K_SZ + kA;

    // B row-of-weight base pointers (per-lane): col n of B = row n of W
    const int col = hn + lrow;
    const _Float16* wih0 = WIH + (size_t)(0 * H_SZ + col) * K_SZ + kB;
    const _Float16* wih1 = WIH + (size_t)(1 * H_SZ + col) * K_SZ + kB;
    const _Float16* wih2 = WIH + (size_t)(2 * H_SZ + col) * K_SZ + kB;
    const _Float16* wih3 = WIH + (size_t)(3 * H_SZ + col) * K_SZ + kB;
    const _Float16* whh0 = WHH + (size_t)(0 * H_SZ + col) * K_SZ + kB;
    const _Float16* whh1 = WHH + (size_t)(1 * H_SZ + col) * K_SZ + kB;
    const _Float16* whh2 = WHH + (size_t)(2 * H_SZ + col) * K_SZ + kB;
    const _Float16* whh3 = WHH + (size_t)(3 * H_SZ + col) * K_SZ + kB;
    const _Float16* wat0 = WAT + (size_t)(0 * H_SZ + col) * K_SZ + kB;
    const _Float16* wat1 = WAT + (size_t)(1 * H_SZ + col) * K_SZ + kB;

    v8f acc[6][2];
    #pragma unroll
    for (int g = 0; g < 6; ++g) { acc[g][0] = (v8f)0.0f; acc[g][1] = (v8f)0.0f; }

    for (int k0 = 0; k0 < K_SZ; k0 += 32) {
        v16h ax0 = loadA(xr0 + k0), ax1 = loadA(xr1 + k0);
        v16h ah0 = loadA(hr0 + k0), ah1 = loadA(hr1 + k0);
        v16h aa0 = loadA(ar0 + k0), aa1 = loadA(ar1 + k0);

        { v16h b = loadB(wih0 + k0);
          acc[0][0] = wmma_f16(ax0, b, acc[0][0]); acc[0][1] = wmma_f16(ax1, b, acc[0][1]); }
        { v16h b = loadB(whh0 + k0);
          acc[0][0] = wmma_f16(ah0, b, acc[0][0]); acc[0][1] = wmma_f16(ah1, b, acc[0][1]); }
        { v16h b = loadB(wih1 + k0);
          acc[1][0] = wmma_f16(ax0, b, acc[1][0]); acc[1][1] = wmma_f16(ax1, b, acc[1][1]); }
        { v16h b = loadB(whh1 + k0);
          acc[1][0] = wmma_f16(ah0, b, acc[1][0]); acc[1][1] = wmma_f16(ah1, b, acc[1][1]); }
        { v16h b = loadB(wih2 + k0);
          acc[2][0] = wmma_f16(ax0, b, acc[2][0]); acc[2][1] = wmma_f16(ax1, b, acc[2][1]); }
        { v16h b = loadB(whh2 + k0);
          acc[2][0] = wmma_f16(ah0, b, acc[2][0]); acc[2][1] = wmma_f16(ah1, b, acc[2][1]); }
        { v16h b = loadB(wih3 + k0);
          acc[3][0] = wmma_f16(ax0, b, acc[3][0]); acc[3][1] = wmma_f16(ax1, b, acc[3][1]); }
        { v16h b = loadB(whh3 + k0);
          acc[3][0] = wmma_f16(ah0, b, acc[3][0]); acc[3][1] = wmma_f16(ah1, b, acc[3][1]); }
        { v16h b = loadB(wat0 + k0);
          acc[4][0] = wmma_f16(aa0, b, acc[4][0]); acc[4][1] = wmma_f16(aa1, b, acc[4][1]); }
        { v16h b = loadB(wat1 + k0);
          acc[5][0] = wmma_f16(aa0, b, acc[5][0]); acc[5][1] = wmma_f16(aa1, b, acc[5][1]); }
    }

    // ---- epilogue: C/D f32 layout: lane -> N=col, VGPR v -> M = v + 8*(lane>=16) ----
    const float b0 = bih[0 * H_SZ + col] + bhh[0 * H_SZ + col];
    const float b1 = bih[1 * H_SZ + col] + bhh[1 * H_SZ + col];
    const float b2 = bih[2 * H_SZ + col] + bhh[2 * H_SZ + col];
    const float b3 = bih[3 * H_SZ + col] + bhh[3 * H_SZ + col];
    const float a0 = batt[col];
    const float a1 = batt[H_SZ + col];

    const int rbase = bm + ((lane >> 4) << 3);
    #pragma unroll
    for (int m = 0; m < 2; ++m) {
        #pragma unroll
        for (int v = 0; v < 8; ++v) {
            const int row = rbase + 16 * m + v;
            float ing  = sigm (acc[0][m][v] + b0);
            float fg   = sigm (acc[1][m][v] + b1);
            float cg   = tanhf(acc[2][m][v] + b2);
            float og   = sigm (acc[3][m][v] + b3);
            float inga = sigm (acc[4][m][v] + a0);
            float atg  = tanhf(acc[5][m][v] + a1);
            float c    = cx[(size_t)row * H_SZ + col];
            float cy   = fg * c + ing * cg + inga * atg;
            float hy   = og * tanhf(cy);
            out[(size_t)row * H_SZ + col] = hy;
            out[(size_t)B_SZ * H_SZ + (size_t)row * H_SZ + col] = cy;
        }
    }
}

// ---------------- host launch ----------------
extern "C" void kernel_launch(void* const* d_in, const int* in_sizes, int n_in,
                              void* d_out, int out_size, void* d_ws, size_t ws_size,
                              hipStream_t stream) {
    const float* input  = (const float*)d_in[0];
    const float* hx     = (const float*)d_in[1];
    const float* cx     = (const float*)d_in[2];
    const float* att    = (const float*)d_in[3];
    const float* w_ih   = (const float*)d_in[4];
    const float* w_hh   = (const float*)d_in[5];
    const float* b_ih   = (const float*)d_in[6];
    const float* b_hh   = (const float*)d_in[7];
    const float* w_att  = (const float*)d_in[8];
    const float* b_att  = (const float*)d_in[9];
    float* out = (float*)d_out;

    const size_t BI = (size_t)B_SZ * K_SZ;     // 8M
    const size_t W4 = (size_t)4 * H_SZ * K_SZ; // 4M
    const size_t W2 = (size_t)2 * H_SZ * K_SZ; // 2M

    _Float16* ws   = (_Float16*)d_ws;
    _Float16* X16  = ws;
    _Float16* HX16 = X16  + BI;
    _Float16* AT16 = HX16 + BI;
    _Float16* WI16 = AT16 + BI;
    _Float16* WH16 = WI16 + W4;
    _Float16* WA16 = WH16 + W4;   // total 34M halves = 68 MB of ws

    auto cvt = [&](const float* s, _Float16* d, size_t n) {
        int n4 = (int)(n / 4);
        cvt_f32_f16<<<(n4 + 255) / 256, 256, 0, stream>>>(s, d, n4);
    };
    cvt(input, X16, BI);
    cvt(hx,    HX16, BI);
    cvt(att,   AT16, BI);
    cvt(w_ih,  WI16, W4);
    cvt(w_hh,  WH16, W4);
    cvt(w_att, WA16, W2);

    // (B/32) * (H/16) = 256*64 = 16384 wave-tiles; 8 waves per 256-thread block
    lstm_att_wmma<<<2048, 256, 0, stream>>>(X16, HX16, AT16, WI16, WH16, WA16,
                                            b_ih, b_hh, b_att, cx, out);
}
